// BALM_MedualTime_57758720197313
// MI455X (gfx1250) — matrix operand order repaired and checked by reading, not compiled, not run on hardware
//
#include <hip/hip_runtime.h>

// Problem constants (match reference)
#define B_    8
#define S_    4096
#define D_    1024
#define M_    32
#define KTOP  3
#define R_    64
#define BS_   (B_ * S_)
#define BSD_  ((size_t)BS_ * (size_t)D_)

// ---------------- WMMA types ----------------
typedef __attribute__((ext_vector_type(16))) __bf16 bf16x16;
typedef __attribute__((ext_vector_type(8)))  float  f32x8;
typedef __attribute__((ext_vector_type(4)))  unsigned int v4u;
typedef __attribute__((ext_vector_type(8)))  int v8i_;
typedef __attribute__((ext_vector_type(4)))  int v4i_;

union Frag16 {
    bf16x16        v;
    unsigned short u[16];
    unsigned int   w[8];
    uint4          q[2];
};

__device__ __forceinline__ unsigned short f2bf(float f) {
    unsigned int x = __float_as_uint(f);
    unsigned int r = (x + 0x7FFFu + ((x >> 16) & 1u)) >> 16;   // RNE
    return (unsigned short)r;
}
__device__ __forceinline__ float bf2f(unsigned short h) {
    return __uint_as_float(((unsigned int)h) << 16);
}

// 256-thread block sum (sm must be float[256])
__device__ __forceinline__ float block_sum(float v, float* sm) {
    int tid = threadIdx.x;
    sm[tid] = v;
    __syncthreads();
    #pragma unroll
    for (int off = 128; off > 0; off >>= 1) {
        if (tid < off) sm[tid] += sm[tid + off];
        __syncthreads();
    }
    float r = sm[0];
    __syncthreads();
    return r;
}

// ---------------- TDM: 2-D tensor tile -> LDS (async, TENSORcnt) ----------------
// group1 word0 flag constants: data_size=2B (code 1) at [17:16], pad_enable bit 20,
// pad_interval [24:22] (code: 1<<code 8-byte units), pad_amount [31:25] (DWORDs-1).
#define TDM_FLAGS_A (0x00010000u | (1u << 20) | (3u << 22) | (3u << 25)) // 64B rows + 16B pad -> stride 40 halves
#define TDM_FLAGS_B (0x00010000u | (1u << 20) | (4u << 22) | (3u << 25)) // 128B rows + 16B pad -> stride 72 halves

__device__ __forceinline__ void tdm_load_2d(
    const void* gptr, const void* lds_ptr, unsigned int flags,
    unsigned int tensor_d0, unsigned int tensor_d1,
    unsigned int tile_d0, unsigned int tile_d1, unsigned int stride0)
{
    unsigned long long ga = (unsigned long long)gptr;
    unsigned int lo = (unsigned int)(unsigned long long)lds_ptr;
    v4u g0;
    g0[0] = 1u;                                                    // count=1, user mode
    g0[1] = lo;                                                    // lds_addr
    g0[2] = (unsigned int)ga;                                      // global_addr[31:0]
    g0[3] = ((unsigned int)(ga >> 32) & 0x1FFFFFFu) | (2u << 30);  // addr[56:32] | type=2
    v8i_ g1;
    g1[0] = (int)flags;
    g1[1] = (int)((tensor_d0 & 0xFFFFu) << 16);                    // tensor_dim0[15:0]
    g1[2] = (int)((tensor_d0 >> 16) | ((tensor_d1 & 0xFFFFu) << 16));
    g1[3] = (int)((tensor_d1 >> 16) | (tile_d0 << 16));            // tile_dim0
    g1[4] = (int)tile_d1;                                          // tile_dim1 (tile_dim2=0)
    g1[5] = (int)stride0;                                          // tensor_dim0_stride[31:0]
    g1[6] = 0;
    g1[7] = 0;
    v4i_ z4; z4[0] = 0; z4[1] = 0; z4[2] = 0; z4[3] = 0;
#if defined(__clang_major__) && (__clang_major__ >= 23)
    v8i_ z8; z8[0]=0; z8[1]=0; z8[2]=0; z8[3]=0; z8[4]=0; z8[5]=0; z8[6]=0; z8[7]=0;
    __builtin_amdgcn_tensor_load_to_lds(g0, g1, z4, z4, z8, 0);
#else
    __builtin_amdgcn_tensor_load_to_lds(g0, g1, z4, z4, 0);
#endif
}

// ---------------- K1: per-batch scores + top-3 ----------------
__global__ __launch_bounds__(256) void balm_topk(
    const float* __restrict__ pt, const float* __restrict__ aw,
    const float* __restrict__ ab, int* __restrict__ aidx)
{
    __shared__ float wsh[D_];
    __shared__ float sv[768];
    __shared__ int   si[768];
    int b = blockIdx.x, tid = threadIdx.x;
    for (int d = tid; d < D_; d += 256) wsh[d] = aw[d];
    __syncthreads();

    float v0 = -1e30f, v1 = -1e30f, v2 = -1e30f;
    int   i0 = -1, i1 = -1, i2 = -1;
    for (int s = tid; s < S_; s += 256) {
        const float* x = pt + ((size_t)b * S_ + s) * D_;
        float sc = ab[0];
        for (int d = 0; d < D_; d++) sc += x[d] * wsh[d];
        if (sc > v0)      { v2 = v1; i2 = i1; v1 = v0; i1 = i0; v0 = sc; i0 = s; }
        else if (sc > v1) { v2 = v1; i2 = i1; v1 = sc; i1 = s; }
        else if (sc > v2) { v2 = sc; i2 = s; }
    }
    sv[tid] = v0;        si[tid] = i0;
    sv[256 + tid] = v1;  si[256 + tid] = i1;
    sv[512 + tid] = v2;  si[512 + tid] = i2;
    __syncthreads();
    if (tid == 0) {
        for (int k = 0; k < KTOP; k++) {
            float best = -1e30f; int bi = 0, bj = -1;
            for (int j = 0; j < 768; j++)
                if (si[j] >= 0 && sv[j] > best) { best = sv[j]; bi = si[j]; bj = j; }
            aidx[b * KTOP + k] = bi;
            if (bj >= 0) { sv[bj] = -1e31f; si[bj] = -1; }
        }
    }
}

// ---------------- K2: per-(b,k) mode-basis translate ----------------
__global__ __launch_bounds__(256) void balm_translate(
    const float* __restrict__ pt, const int* __restrict__ aidx,
    const float* __restrict__ cent, const float* __restrict__ tra,
    const float* __restrict__ trb, float* __restrict__ transl)
{
    __shared__ float red[256];
    __shared__ float a[D_];
    __shared__ float wgt[M_];
    __shared__ float lg[M_];
    __shared__ float lowr[R_];
    int bk = blockIdx.x;
    int b  = bk / KTOP;
    int tid = threadIdx.x;
    int idx = aidx[bk];
    for (int d = tid; d < D_; d += 256)
        a[d] = pt[((size_t)b * S_ + idx) * D_ + d];
    __syncthreads();

    float pa = 0.f;
    for (int d = tid; d < D_; d += 256) pa += a[d] * a[d];
    float na   = sqrtf(block_sum(pa, red));
    float inva = 1.f / fmaxf(na, 1e-6f);

    for (int m = 0; m < M_; m++) {
        float pd = 0.f, pc = 0.f;
        for (int d = tid; d < D_; d += 256) {
            float c = cent[(size_t)m * D_ + d];
            pd += a[d] * c;
            pc += c * c;
        }
        float sd = block_sum(pd, red);
        float sc = block_sum(pc, red);
        if (tid == 0) lg[m] = (sd * inva) / fmaxf(sqrtf(sc), 1e-6f);
    }
    __syncthreads();
    if (tid == 0) {
        float mx = -1e30f;
        for (int m = 0; m < M_; m++) mx = fmaxf(mx, lg[m]);
        float sum = 0.f;
        for (int m = 0; m < M_; m++) { float e = __expf(lg[m] - mx); wgt[m] = e; sum += e; }
        float isum = 1.f / sum;
        for (int m = 0; m < M_; m++) wgt[m] *= isum;
    }
    __syncthreads();

    // low[r] = sum_d a[d] * (sum_m w[m]*trans_a[m,d,r])
    int r = tid & 63, grp = tid >> 6;
    float p = 0.f;
    for (int d = grp; d < D_; d += 4) {
        float ws = 0.f;
        #pragma unroll 8
        for (int m = 0; m < M_; m++) ws += wgt[m] * tra[((size_t)m * D_ + d) * R_ + r];
        p += a[d] * ws;
    }
    red[tid] = p;
    __syncthreads();
    if (tid < 64) lowr[tid] = red[tid] + red[tid + 64] + red[tid + 128] + red[tid + 192];
    __syncthreads();

    // translated[d] = sum_m w[m] * (sum_r low[r]*trans_b[m,d,r])
    for (int d = tid; d < D_; d += 256) {
        float acc = 0.f;
        for (int m = 0; m < M_; m++) {
            const float* tb = &trb[((size_t)m * D_ + d) * R_];
            float dr = 0.f;
            #pragma unroll 8
            for (int rr = 0; rr < R_; rr++) dr += lowr[rr] * tb[rr];
            acc += wgt[m] * dr;
        }
        transl[(size_t)bk * D_ + d] = acc;
    }
}

// ---------------- K3: per-token anchor spread + 3x LN + cos gate ----------------
__global__ __launch_bounds__(256) void balm_token(
    const float* __restrict__ pt, const float* __restrict__ ts,
    const float* __restrict__ pos, const float* __restrict__ stab,
    const int* __restrict__ aidx, const float* __restrict__ transl,
    const float* __restrict__ og, const float* __restrict__ ob,
    const float* __restrict__ tg, const float* __restrict__ tbv,
    const float* __restrict__ pg, const float* __restrict__ pb,
    float* __restrict__ ts_aug, unsigned short* __restrict__ A1,
    float* __restrict__ gate)
{
    __shared__ float red[256];
    __shared__ float wk[KTOP];
    int t = blockIdx.x;
    int b = t >> 12;          // / S_
    int s = t & (S_ - 1);
    int tid = threadIdx.x;

    if (tid < KTOP) {
        int idx = aidx[b * KTOP + tid];
        float p = pos[idx];
        int pi = (int)p;
        int lo = pi - 8; if (lo < 0) lo = 0;
        int hi = pi + 8; if (hi > S_ - 1) hi = S_ - 1;
        float Z = 0.f;
        for (int q = lo; q <= hi; q++) Z += __expf(-fabsf(pos[q] - p) * 0.125f);
        float dist = fabsf(pos[s] - p);
        wk[tid] = (dist <= 8.f) ? __expf(-dist * 0.125f) / Z : 0.f;
    }
    __syncthreads();
    float w0 = wk[0], w1 = wk[1], w2 = wk[2];

    float u[4], pv[4];
    #pragma unroll
    for (int e = 0; e < 4; e++) {
        int d = tid + 256 * e;
        float upd = w0 * transl[(size_t)(b * KTOP + 0) * D_ + d]
                  + w1 * transl[(size_t)(b * KTOP + 1) * D_ + d]
                  + w2 * transl[(size_t)(b * KTOP + 2) * D_ + d];
        u[e]  = ts[(size_t)t * D_ + d] + upd;
        pv[e] = pt[(size_t)t * D_ + d];
    }

    // LN(u) with out_norm -> ts_aug
    float sm = u[0] + u[1] + u[2] + u[3];
    float mu = block_sum(sm, red) * (1.f / D_);
    float sv = 0.f;
    #pragma unroll
    for (int e = 0; e < 4; e++) { float c = u[e] - mu; sv += c * c; }
    float inv = rsqrtf(block_sum(sv, red) * (1.f / D_) + 1e-5f);
    float ta[4];
    #pragma unroll
    for (int e = 0; e < 4; e++) {
        int d = tid + 256 * e;
        ta[e] = (u[e] - mu) * inv * og[d] + ob[d];
        ts_aug[(size_t)t * D_ + d] = ta[e];
    }

    // tsn = LN(ts_aug) with ts_norm
    sm = ta[0] + ta[1] + ta[2] + ta[3];
    float mu2 = block_sum(sm, red) * (1.f / D_);
    sv = 0.f;
    #pragma unroll
    for (int e = 0; e < 4; e++) { float c = ta[e] - mu2; sv += c * c; }
    float inv2 = rsqrtf(block_sum(sv, red) * (1.f / D_) + 1e-5f);
    float tn[4];
    #pragma unroll
    for (int e = 0; e < 4; e++) {
        int d = tid + 256 * e;
        tn[e] = (ta[e] - mu2) * inv2 * tg[d] + tbv[d];
    }

    // ptn = LN(pt) with pt_norm
    sm = pv[0] + pv[1] + pv[2] + pv[3];
    float mu3 = block_sum(sm, red) * (1.f / D_);
    sv = 0.f;
    #pragma unroll
    for (int e = 0; e < 4; e++) { float c = pv[e] - mu3; sv += c * c; }
    float inv3 = rsqrtf(block_sum(sv, red) * (1.f / D_) + 1e-5f);
    float pn[4];
    #pragma unroll
    for (int e = 0; e < 4; e++) {
        int d = tid + 256 * e;
        pn[e] = (pv[e] - mu3) * inv3 * pg[d] + pb[d];
    }

    // cosine / agreement / focus gate
    float dd = 0.f, q1 = 0.f, q2 = 0.f;
    #pragma unroll
    for (int e = 0; e < 4; e++) { dd += tn[e] * pn[e]; q1 += tn[e] * tn[e]; q2 += pn[e] * pn[e]; }
    dd = block_sum(dd, red);
    q1 = block_sum(q1, red);
    q2 = block_sum(q2, red);
    float cosv  = dd / (fmaxf(sqrtf(q1), 1e-6f) * fmaxf(sqrtf(q2), 1e-6f));
    float agr   = 0.5f * (1.f + cosv);
    float z     = (agr - 0.72f) * 5.f;
    float focus = 0.2f + 0.8f * __expf(-0.5f * z * z);
    if (tid == 0) gate[t] = agr * focus * stab[t];

    // bf16 [tsn | ptn] activation row, K = 2048
    #pragma unroll
    for (int e = 0; e < 4; e++) {
        int d = tid + 256 * e;
        A1[(size_t)t * 2048 + d]       = f2bf(tn[e]);
        A1[(size_t)t * 2048 + D_ + d]  = f2bf(pn[e]);
    }
}

// ---------------- K4: f32 -> bf16 weight conversion ----------------
__global__ __launch_bounds__(256) void balm_convert(
    const float* __restrict__ sW, const float* __restrict__ tW,
    const float* __restrict__ pW, unsigned short* __restrict__ wsh,
    unsigned short* __restrict__ wts, unsigned short* __restrict__ wpt)
{
    const size_t N1 = (size_t)2048 * 1024, N2 = (size_t)1024 * 1024;
    size_t stride = (size_t)gridDim.x * blockDim.x;
    for (size_t i = (size_t)blockIdx.x * blockDim.x + threadIdx.x;
         i < N1 + 2 * N2; i += stride) {
        if (i < N1)            wsh[i]            = f2bf(sW[i]);
        else if (i < N1 + N2)  wts[i - N1]       = f2bf(tW[i - N1]);
        else                   wpt[i - N1 - N2]  = f2bf(pW[i - N1 - N2]);
    }
}

// ---------------- K5/K7: WMMA bf16 GEMM with TDM double-buffered staging ----------------
// Block tile 128x64, 8 waves, wave tile 32x32 (2x2 x v_wmma_f32_16x16x32_bf16).
// A/B tiles are DMA'd into LDS by the Tensor Data Mover (pad feature supplies the
// LDS row padding: A 64B rows + 16B pad -> 40-half stride; B 128B + 16B -> 72).
// mode 0: out = acc + gate[m]*wlast[n] + bias[n]
// mode 1: out = base + 0.35*sigmoid(*bscale)*gate[m]*acc
#define ASH_   (128 * 40)          // A-tile halves (incl. pad)
#define BSH_   (32 * 72)           // B-tile halves (incl. pad)
#define BUFH_  (ASH_ + BSH_)       // halves per pipeline buffer

__global__ __launch_bounds__(256) void balm_gemm(
    const unsigned short* __restrict__ A, int lda,
    const unsigned short* __restrict__ Bw, int Kdim,
    float* __restrict__ out, const float* __restrict__ base,
    const float* __restrict__ gate, const float* __restrict__ wlast,
    const float* __restrict__ bias, const float* __restrict__ bscale,
    int mode)
{
    extern __shared__ __align__(16) unsigned short dynlds[];
    unsigned short* buf0 = dynlds;
    unsigned short* buf1 = dynlds + BUFH_;

    int tid  = threadIdx.x;
    int m0   = blockIdx.y * 128;
    int n0   = blockIdx.x * 64;
    int lane = tid & 31;
    int wave = tid >> 5;
    int wm   = (wave >> 1) * 32;   // 0/32/64/96
    int wn   = (wave & 1) * 32;    // 0/32

    f32x8 acc[2][2];
    #pragma unroll
    for (int i = 0; i < 2; i++)
        #pragma unroll
        for (int j = 0; j < 2; j++)
            #pragma unroll
            for (int v = 0; v < 8; v++) acc[i][j][v] = 0.f;

    // prologue: DMA first K-tile into buffer 0
    if (tid == 0) {
        tdm_load_2d(A + (size_t)m0 * lda, buf0, TDM_FLAGS_A,
                    (unsigned)lda, 32768u, 32u, 128u, (unsigned)lda);
        tdm_load_2d(Bw + n0, buf0 + ASH_, TDM_FLAGS_B,
                    1024u, (unsigned)Kdim, 64u, 32u, 1024u);
    }
    __builtin_amdgcn_s_wait_tensorcnt(0);
    __syncthreads();

    for (int kt = 0; kt < Kdim; kt += 32) {
        int pbuf = (kt >> 5) & 1;
        const unsigned short* Acur = pbuf ? buf1 : buf0;
        const unsigned short* Bcur = Acur + ASH_;

        if (kt + 32 < Kdim) {
            if (tid == 0) {   // one TDM issuer per workgroup; DMA overlaps the WMMAs below
                unsigned short* nb = pbuf ? buf0 : buf1;
                tdm_load_2d(A + (size_t)m0 * lda + (kt + 32), nb, TDM_FLAGS_A,
                            (unsigned)lda, 32768u, 32u, 128u, (unsigned)lda);
                tdm_load_2d(Bw + (size_t)(kt + 32) * 1024 + n0, nb + ASH_, TDM_FLAGS_B,
                            1024u, (unsigned)Kdim, 64u, 32u, 1024u);
            }
            if (wave == 0 && lane < 8)   // warm L2 for the next K-tile, the one the DMA above fetches
                __builtin_prefetch(A + (size_t)(m0 + lane * 16) * lda + kt + 32, 0, 1);
        }

        // A fragments: 16x32 bf16; lane%16 = M row, lane/16 selects K group
        Frag16 af[2], bfr[2];
        int kg = lane >> 4;
        #pragma unroll
        for (int i = 0; i < 2; i++) {
            int row = wm + 16 * i + (lane & 15);
            const unsigned short* asrc = &Acur[row * 40];
            #pragma unroll
            for (int r = 0; r < 8; r++) {
                int k0 = ((r < 4) ? 0 : 16) + kg * 8 + (r & 3) * 2;
                af[i].w[r] = *(const unsigned int*)(&asrc[k0]);
            }
        }
        // B fragments: 32x16 bf16; lane%16 = N col, lane/16 selects K group
        #pragma unroll
        for (int j = 0; j < 2; j++) {
            const unsigned short* bsrc = &Bcur[wn + 16 * j + (lane & 15)];
            #pragma unroll
            for (int s = 0; s < 16; s++) {
                int k = ((s < 8) ? 0 : 16) + kg * 8 + (s & 7);
                bfr[j].u[s] = bsrc[k * 72];
            }
        }
        #pragma unroll
        for (int i = 0; i < 2; i++)
            #pragma unroll
            for (int j = 0; j < 2; j++)
                acc[i][j] = __builtin_amdgcn_wmma_f32_16x16x32_bf16(
                    false, af[i].v, false, bfr[j].v, (short)0, acc[i][j],
                    false, false);

        __builtin_amdgcn_s_wait_tensorcnt(0);   // issuer waits for next tile; others no-op
        __syncthreads();
    }

    float blend = 0.f;
    if (mode == 1) blend = 0.35f * (1.0f / (1.0f + __expf(-bscale[0])));
    int nlane = lane & 15;
    int mofs  = (lane >> 4) * 8;
    #pragma unroll
    for (int i = 0; i < 2; i++) {
        #pragma unroll
        for (int j = 0; j < 2; j++) {
            #pragma unroll
            for (int v = 0; v < 8; v++) {
                int mrow = m0 + wm + 16 * i + mofs + v;
                int ncol = n0 + wn + 16 * j + nlane;
                size_t o = (size_t)mrow * 1024 + ncol;
                float a = acc[i][j][v];
                if (mode == 0) {
                    out[o] = a + gate[mrow] * wlast[ncol] + bias[ncol];
                } else {
                    out[o] = base[o] + blend * gate[mrow] * a;
                }
            }
        }
    }
}

// ---------------- K6: causal pool (k=3) + subtract tsn/ptn -> bf16 operands ----------------
__global__ __launch_bounds__(256) void balm_pool(
    const float* __restrict__ sh, const unsigned short* __restrict__ A1,
    unsigned short* __restrict__ Ats, unsigned short* __restrict__ Apt)
{
    size_t stride = (size_t)gridDim.x * blockDim.x;
    for (size_t i = (size_t)blockIdx.x * blockDim.x + threadIdx.x;
         i < BSD_; i += stride) {
        size_t t = i >> 10;
        int n = (int)(i & 1023);
        int s = (int)(t & (S_ - 1));
        size_t bt = t - s;
        int s1 = (s >= 1) ? s - 1 : 0;
        int s2 = (s >= 2) ? s - 2 : 0;
        float p = (sh[i] + sh[(bt + s1) * 1024 + n] + sh[(bt + s2) * 1024 + n]) * (1.f / 3.f);
        float tn = bf2f(A1[t * 2048 + n]);
        float pn = bf2f(A1[t * 2048 + 1024 + n]);
        Ats[i] = f2bf(p - tn);
        Apt[i] = f2bf(p - pn);
    }
}

// ---------------- workspace layout ----------------
static const size_t OFF_TSAUG  = 0;                                       // f32 [BS,1024]
static const size_t OFF_A1     = OFF_TSAUG  + BSD_ * 4;                   // bf16 [BS,2048]
static const size_t OFF_SHARED = OFF_A1     + (size_t)BS_ * 2048 * 2;     // f32 [BS,1024]
static const size_t OFF_ATS    = OFF_SHARED + BSD_ * 4;                   // bf16 [BS,1024]
static const size_t OFF_APT    = OFF_ATS    + BSD_ * 2;                   // bf16 [BS,1024]
static const size_t OFF_WSH    = OFF_APT    + BSD_ * 2;                   // bf16 [2048,1024]
static const size_t OFF_WTS    = OFF_WSH    + (size_t)2048 * 1024 * 2;    // bf16 [1024,1024]
static const size_t OFF_WPT    = OFF_WTS    + (size_t)1024 * 1024 * 2;    // bf16 [1024,1024]
static const size_t OFF_TRANS  = OFF_WPT    + (size_t)1024 * 1024 * 2;    // f32 [24,1024]
static const size_t OFF_GATE   = OFF_TRANS  + (size_t)B_ * KTOP * D_ * 4; // f32 [BS]
static const size_t OFF_IDX    = OFF_GATE   + (size_t)BS_ * 4;            // int [24]

extern "C" void kernel_launch(void* const* d_in, const int* in_sizes, int n_in,
                              void* d_out, int out_size, void* d_ws, size_t ws_size,
                              hipStream_t stream)
{
    const float* pt   = (const float*)d_in[0];
    const float* ts   = (const float*)d_in[1];
    const float* pos  = (const float*)d_in[2];
    const float* cent = (const float*)d_in[3];
    const float* stab = (const float*)d_in[4];
    const float* aw   = (const float*)d_in[5];
    const float* ab   = (const float*)d_in[6];
    const float* tra  = (const float*)d_in[7];
    const float* trb  = (const float*)d_in[8];
    const float* og   = (const float*)d_in[9];
    const float* ob   = (const float*)d_in[10];
    const float* tg   = (const float*)d_in[11];
    const float* tbv  = (const float*)d_in[12];
    const float* pg   = (const float*)d_in[13];
    const float* pb   = (const float*)d_in[14];
    const float* sW   = (const float*)d_in[15];
    const float* sb   = (const float*)d_in[16];
    const float* tuW  = (const float*)d_in[17];
    const float* puW  = (const float*)d_in[18];
    const float* bsc  = (const float*)d_in[19];

    unsigned char* ws = (unsigned char*)d_ws;
    float*          ts_aug = (float*)(ws + OFF_TSAUG);
    unsigned short* A1     = (unsigned short*)(ws + OFF_A1);
    float*          shmat  = (float*)(ws + OFF_SHARED);
    unsigned short* Ats    = (unsigned short*)(ws + OFF_ATS);
    unsigned short* Apt    = (unsigned short*)(ws + OFF_APT);
    unsigned short* Wsh    = (unsigned short*)(ws + OFF_WSH);
    unsigned short* Wts    = (unsigned short*)(ws + OFF_WTS);
    unsigned short* Wpt    = (unsigned short*)(ws + OFF_WPT);
    float*          transl = (float*)(ws + OFF_TRANS);
    float*          gate   = (float*)(ws + OFF_GATE);
    int*            aidx   = (int*)(ws + OFF_IDX);

    float* ts_out = (float*)d_out;
    float* pt_out = ts_out + BSD_;

    const int gemm_lds = (int)(2 * BUFH_ * sizeof(unsigned short));  // double-buffered tiles

    balm_topk<<<B_, 256, 0, stream>>>(pt, aw, ab, aidx);
    balm_translate<<<B_ * KTOP, 256, 0, stream>>>(pt, aidx, cent, tra, trb, transl);
    balm_token<<<BS_, 256, 0, stream>>>(pt, ts, pos, stab, aidx, transl,
                                        og, ob, tg, tbv, pg, pb,
                                        ts_aug, A1, gate);
    balm_convert<<<4096, 256, 0, stream>>>(sW, tuW, puW, Wsh, Wts, Wpt);

    // shared = [tsn|ptn] @ Wsh  (+ gate * W[2048,:] + bias)   M=32768 N=1024 K=2048
    balm_gemm<<<dim3(16, 256), 256, gemm_lds, stream>>>(
        A1, 2048, Wsh, 2048, shmat, nullptr, gate,
        sW + (size_t)2048 * 1024, sb, bsc, 0);

    balm_pool<<<8192, 256, 0, stream>>>(shmat, A1, Ats, Apt);

    // ts_out = ts_aug + lb * ((pool(shared) - tsn) @ ts_up_W)
    balm_gemm<<<dim3(16, 256), 256, gemm_lds, stream>>>(
        Ats, 1024, Wts, 1024, ts_out, ts_aug, gate, nullptr, nullptr, bsc, 1);
    // pt_out = pt_hidden + lb * ((pool(shared) - ptn) @ pt_up_W)
    balm_gemm<<<dim3(16, 256), 256, gemm_lds, stream>>>(
        Apt, 1024, Wpt, 1024, pt_out, pt, gate, nullptr, nullptr, bsc, 1);
}
